// LineAttention_55044300865546
// MI455X (gfx1250) — matrix operand order repaired and checked by reading, not compiled, not run on hardware
//
#include <hip/hip_runtime.h>

typedef __bf16 bf16_t;
typedef __attribute__((ext_vector_type(8)))  bf16_t v8bf;
typedef __attribute__((ext_vector_type(16))) bf16_t v16bf;
typedef __attribute__((ext_vector_type(8)))  float  v8f;

static __device__ inline bf16_t f2bf(float f) {
  unsigned u = __builtin_bit_cast(unsigned, f);
  u += 0x7FFFu + ((u >> 16) & 1u);          // round-to-nearest-even
  unsigned short h = (unsigned short)(u >> 16);
  return __builtin_bit_cast(bf16_t, h);
}

static __device__ inline v16bf join8(v8bf lo, v8bf hi) {
  return __builtin_shufflevector(lo, hi, 0,1,2,3,4,5,6,7,8,9,10,11,12,13,14,15);
}

// A-fragment (16x32, M=row): lane<16 holds K {0..7,16..23}, lane>=16 K {8..15,24..31}.
static __device__ inline v16bf load_a(const bf16_t* p, int ld, int lane) {
  int half = lane >> 4, row = lane & 15;
  const bf16_t* q = p + row * ld + half * 8;
  v8bf lo = *(const v8bf*)(q);
  v8bf hi = *(const v8bf*)(q + 16);
  return join8(lo, hi);
}

// B-fragment (32x16) from K-contiguous layout Bt[n][k]: lane<16 holds K {0..7,16..23}, lane>=16 K {8..15,24..31}.
static __device__ inline v16bf load_bt(const bf16_t* p, int ld, int lane) {
  int half = lane >> 4, col = lane & 15;
  const bf16_t* q = p + col * ld + half * 8;
  v8bf lo = *(const v8bf*)(q);
  v8bf hi = *(const v8bf*)(q + 16);
  return join8(lo, hi);
}

static __device__ inline v8f wmma_bf16(v16bf a, v16bf b, v8f c) {
  return __builtin_amdgcn_wmma_f32_16x16x32_bf16(false, a, false, b, (short)0, c, false, false);
}

#define EMB 1024
#define HEADS 16
#define HDIM 64
#define BATCH 4
#define SEQ 1024
#define MROWS (BATCH * SEQ)

// ping-pong GEMM inner-loop helpers (true double buffering, no register copies)
#define LOADSET(WP, S, KK)                                                      \
  do {                                                                          \
    aA[S][0] = load_a(pa0 + (KK), EMB, lane);                                   \
    aA[S][1] = load_a(pa1 + (KK), EMB, lane);                                   \
    _Pragma("unroll")                                                           \
    for (int t_ = 0; t_ < 4; ++t_)                                              \
      bB[S][t_] = load_bt((WP) + (n0 + t_ * 16) * EMB + (KK), EMB, lane);       \
  } while (0)

#define COMPUTE(S)                                                              \
  do {                                                                          \
    _Pragma("unroll")                                                           \
    for (int t_ = 0; t_ < 4; ++t_) {                                            \
      acc[0][t_] = wmma_bf16(aA[S][0], bB[S][t_], acc[0][t_]);                  \
      acc[1][t_] = wmma_bf16(aA[S][1], bB[S][t_], acc[1][t_]);                  \
    }                                                                           \
  } while (0)

// ---------------- prep kernels ----------------
__global__ void k_cvt_x(const float* __restrict__ x, bf16_t* __restrict__ xb, int n) {
  int i = blockIdx.x * blockDim.x + threadIdx.x;
  if (i < n) xb[i] = f2bf(x[i]);
}

__global__ void k_tr_w(const float* __restrict__ W, bf16_t* __restrict__ Wt) {
  int i = blockIdx.x * blockDim.x + threadIdx.x;   // 1M elements
  int k = i >> 10, n = i & 1023;
  Wt[n * EMB + k] = f2bf(W[i]);
}

// ---------------- fused QKV projection ----------------
// grid (MROWS/128, 48), 128 threads (4 waves, each a 32x64 tile, ping-pong pipelined).
__global__ __launch_bounds__(128) void k_qkv(
    const bf16_t* __restrict__ xb,
    const bf16_t* __restrict__ wtq, const bf16_t* __restrict__ wtk, const bf16_t* __restrict__ wtv,
    const float* __restrict__ bq, const float* __restrict__ bk, const float* __restrict__ bv,
    bf16_t* __restrict__ Q, bf16_t* __restrict__ K, bf16_t* __restrict__ Vt) {
  int lane = threadIdx.x & 31;
  int wave = threadIdx.x >> 5;
  int m0 = blockIdx.x * 128 + wave * 32;
  int ny = blockIdx.y;
  int mat = ny >> 4;
  int n0 = (ny & 15) << 6;
  const bf16_t* wt   = (mat == 0) ? wtq : (mat == 1) ? wtk : wtv;
  const float*  bias = (mat == 0) ? bq  : (mat == 1) ? bk  : bv;

  const bf16_t* pa0 = xb + (size_t)m0 * EMB;
  const bf16_t* pa1 = pa0 + 16 * EMB;

  v8f acc[2][4] = {};
  v16bf aA[2][2];
  v16bf bB[2][4];

  LOADSET(wt, 0, 0);
  int k = 0;
  for (; k + 64 < EMB; k += 64) {        // 15 double-steps
    LOADSET(wt, 1, k + 32);
    COMPUTE(0);
    LOADSET(wt, 0, k + 64);
    COMPUTE(1);
  }
  LOADSET(wt, 1, k + 32);                // tail: k = 960 / 992
  COMPUTE(0);
  COMPUTE(1);

  int half = lane >> 4, nl = lane & 15;
#pragma unroll
  for (int t = 0; t < 4; ++t) {
    int n = n0 + t * 16 + nl;
    float bb = bias[n];
    int h = n >> 6, d = n & 63;
#pragma unroll
    for (int mi = 0; mi < 2; ++mi)
#pragma unroll
      for (int r = 0; r < 8; ++r) {
        int m = m0 + mi * 16 + r + half * 8;
        int b_ = m >> 10, s = m & 1023;
        bf16_t o = f2bf(acc[mi][t][r] + bb);
        if (mat == 2)      Vt[((b_ * HEADS + h) * HDIM + d) * SEQ + s] = o;   // [b,h,d,s]
        else if (mat == 0) Q [((b_ * HEADS + h) * SEQ + s) * HDIM + d] = o;
        else               K [((b_ * HEADS + h) * SEQ + s) * HDIM + d] = o;
      }
  }
}

// ---------------- flash attention ----------------
// 1024 blocks of 4 waves; block covers 64 queries, wave w handles 16.
// K/V key-tiles (32 keys) staged in LDS once per block -> 4x less L2 traffic.
__global__ __launch_bounds__(128) void k_attn(
    const bf16_t* __restrict__ Q, const bf16_t* __restrict__ K,
    const bf16_t* __restrict__ Vt, bf16_t* __restrict__ O) {
  __shared__ __align__(16) bf16_t sK[32 * 64];      // [key][d]
  __shared__ __align__(16) bf16_t sV[64 * 32];      // [d][key]
  __shared__ __align__(16) bf16_t sP[4][16 * 32];   // per-wave P transpose buffer

  int tid  = threadIdx.x;
  int lane = tid & 31;
  int wave = tid >> 5;
  int bid  = blockIdx.x;
  int qg = bid & 15, h = (bid >> 4) & 15, b = bid >> 8;
  int q0 = qg << 6;             // block's 64-query base
  int qw = q0 + wave * 16;      // this wave's query base
  int half = lane >> 4, nl = lane & 15;

  const bf16_t* Qb = Q  + (size_t)(b * HEADS + h) * SEQ * HDIM;
  const bf16_t* Kb = K  + (size_t)(b * HEADS + h) * SEQ * HDIM;
  const bf16_t* Vb = Vt + (size_t)(b * HEADS + h) * HDIM * SEQ;

  v16bf aq0 = load_a(Qb + qw * HDIM + 0,  HDIM, lane);
  v16bf aq1 = load_a(Qb + qw * HDIM + 32, HDIM, lane);

  v8f accO[4] = {};
  float mrun[8], lrun[8];
#pragma unroll
  for (int r = 0; r < 8; ++r) { mrun[r] = -3.0e38f; lrun[r] = 0.0f; }

  int ktiles = (q0 + 95) >> 5;  // cover keys for the block's last query
  for (int kt = 0; kt < ktiles; ++kt) {
    int kb = kt << 5;
    __syncthreads();            // previous tile fully consumed
    {   // cooperative staging: 128 threads, 16 bf16 each per buffer
      int key = tid >> 2, dc = (tid & 3) * 16;
      const v8bf* srcK = (const v8bf*)(Kb + (kb + key) * HDIM + dc);
      *(v8bf*)(&sK[key * 64 + dc])     = srcK[0];
      *(v8bf*)(&sK[key * 64 + dc + 8]) = srcK[1];
      int d = tid >> 1, kc = (tid & 1) * 16;
      const v8bf* srcV = (const v8bf*)(Vb + d * SEQ + kb + kc);
      *(v8bf*)(&sV[d * 32 + kc])     = srcV[0];
      *(v8bf*)(&sV[d * 32 + kc + 8]) = srcV[1];
    }
    __syncthreads();

    // ---- scores S = Q K^T (16x32) from LDS ----
    v8f sc[2] = {};
#pragma unroll
    for (int t = 0; t < 2; ++t) {
      v16bf bk0 = load_bt(&sK[(t * 16) * 64 + 0],  64, lane);
      v16bf bk1 = load_bt(&sK[(t * 16) * 64 + 32], 64, lane);
      sc[t] = wmma_bf16(aq0, bk0, sc[t]);
      sc[t] = wmma_bf16(aq1, bk1, sc[t]);
    }
    // ---- scale + causal mask ----
#pragma unroll
    for (int t = 0; t < 2; ++t)
#pragma unroll
      for (int r = 0; r < 8; ++r) {
        int qi  = qw + r + half * 8;
        int key = kb + t * 16 + nl;
        float v = sc[t][r] * 0.125f;
        sc[t][r] = (key <= qi) ? v : -3.0e38f;
      }
    // ---- online softmax ----
    float alpha[8];
#pragma unroll
    for (int r = 0; r < 8; ++r) {
      float mx = fmaxf(sc[0][r], sc[1][r]);
#pragma unroll
      for (int m = 1; m <= 8; m <<= 1) mx = fmaxf(mx, __shfl_xor(mx, m, 32));
      float mnew = fmaxf(mrun[r], mx);
      alpha[r] = __expf(mrun[r] - mnew);
      float p0 = __expf(sc[0][r] - mnew);
      float p1 = __expf(sc[1][r] - mnew);
      sc[0][r] = p0; sc[1][r] = p1;
      float rs = p0 + p1;
#pragma unroll
      for (int m = 1; m <= 8; m <<= 1) rs += __shfl_xor(rs, m, 32);
      lrun[r] = alpha[r] * lrun[r] + rs;
      mrun[r] = mnew;
    }
    // ---- P: C-layout -> A-layout via per-wave LDS buffer ----
#pragma unroll
    for (int t = 0; t < 2; ++t)
#pragma unroll
      for (int r = 0; r < 8; ++r)
        sP[wave][(r + half * 8) * 32 + t * 16 + nl] = f2bf(sc[t][r]);
    asm volatile("s_wait_dscnt 0" ::: "memory");
    v8bf plo = *(const v8bf*)&sP[wave][nl * 32 + half * 8];
    v8bf phi = *(const v8bf*)&sP[wave][nl * 32 + 16 + half * 8];
    v16bf aP = join8(plo, phi);
    // ---- O = diag(alpha) O + P V ----
#pragma unroll
    for (int c = 0; c < 4; ++c) {
#pragma unroll
      for (int r = 0; r < 8; ++r) accO[c][r] *= alpha[r];
      v16bf bv = load_bt(&sV[(c * 16) * 32], 32, lane);
      accO[c] = wmma_bf16(aP, bv, accO[c]);
    }
  }
  // ---- normalize + store O[b,s,e] bf16 ----
#pragma unroll
  for (int c = 0; c < 4; ++c)
#pragma unroll
    for (int r = 0; r < 8; ++r) {
      int s = qw + r + half * 8;
      int d = h * HDIM + c * 16 + nl;
      O[(size_t)(b * SEQ + s) * EMB + d] = f2bf(accO[c][r] / lrun[r]);
    }
}

// ---------------- output projection (32x64 per wave, ping-pong pipelined) ----------------
__global__ __launch_bounds__(128) void k_oproj(
    const bf16_t* __restrict__ O, const bf16_t* __restrict__ wto,
    const float* __restrict__ bo, float* __restrict__ out) {
  int lane = threadIdx.x & 31;
  int wave = threadIdx.x >> 5;
  int m0 = blockIdx.x * 128 + wave * 32;
  int n0 = blockIdx.y * 64;

  const bf16_t* pa0 = O + (size_t)m0 * EMB;
  const bf16_t* pa1 = pa0 + 16 * EMB;

  v8f acc[2][4] = {};
  v16bf aA[2][2];
  v16bf bB[2][4];

  LOADSET(wto, 0, 0);
  int k = 0;
  for (; k + 64 < EMB; k += 64) {
    LOADSET(wto, 1, k + 32);
    COMPUTE(0);
    LOADSET(wto, 0, k + 64);
    COMPUTE(1);
  }
  LOADSET(wto, 1, k + 32);
  COMPUTE(0);
  COMPUTE(1);

  int half = lane >> 4, nl = lane & 15;
#pragma unroll
  for (int t = 0; t < 4; ++t) {
    int n = n0 + t * 16 + nl;
    float bb = bo[n];
#pragma unroll
    for (int mi = 0; mi < 2; ++mi)
#pragma unroll
      for (int r = 0; r < 8; ++r) {
        int m = m0 + mi * 16 + r + half * 8;
        out[(size_t)m * EMB + n] = acc[mi][t][r] + bb;
      }
  }
}

extern "C" void kernel_launch(void* const* d_in, const int* in_sizes, int n_in,
                              void* d_out, int out_size, void* d_ws, size_t ws_size,
                              hipStream_t stream) {
  const float* x  = (const float*)d_in[0];
  const float* Wq = (const float*)d_in[1];
  const float* bq = (const float*)d_in[2];
  const float* Wk = (const float*)d_in[3];
  const float* bk = (const float*)d_in[4];
  const float* Wv = (const float*)d_in[5];
  const float* bv = (const float*)d_in[6];
  const float* Wo = (const float*)d_in[7];
  const float* bo = (const float*)d_in[8];
  float* out = (float*)d_out;

  char* ws = (char*)d_ws;
  size_t off = 0;
  bf16_t* xb  = (bf16_t*)(ws + off); off += (size_t)MROWS * EMB * 2;
  bf16_t* wtq = (bf16_t*)(ws + off); off += (size_t)EMB * EMB * 2;
  bf16_t* wtk = (bf16_t*)(ws + off); off += (size_t)EMB * EMB * 2;
  bf16_t* wtv = (bf16_t*)(ws + off); off += (size_t)EMB * EMB * 2;
  bf16_t* wto = (bf16_t*)(ws + off); off += (size_t)EMB * EMB * 2;
  bf16_t* Qb  = (bf16_t*)(ws + off); off += (size_t)MROWS * EMB * 2;
  bf16_t* Kb  = (bf16_t*)(ws + off); off += (size_t)MROWS * EMB * 2;
  bf16_t* Vtb = (bf16_t*)(ws + off); off += (size_t)MROWS * EMB * 2;
  bf16_t* Ob  = (bf16_t*)(ws + off); off += (size_t)MROWS * EMB * 2;   // 48 MB total

  k_cvt_x<<<(MROWS * EMB) / 256, 256, 0, stream>>>(x, xb, MROWS * EMB);
  k_tr_w<<<(EMB * EMB) / 256, 256, 0, stream>>>(Wq, wtq);
  k_tr_w<<<(EMB * EMB) / 256, 256, 0, stream>>>(Wk, wtk);
  k_tr_w<<<(EMB * EMB) / 256, 256, 0, stream>>>(Wv, wtv);
  k_tr_w<<<(EMB * EMB) / 256, 256, 0, stream>>>(Wo, wto);

  k_qkv<<<dim3(MROWS / 128, 48), 128, 0, stream>>>(xb, wtq, wtk, wtv, bq, bk, bv, Qb, Kb, Vtb);
  k_attn<<<BATCH * HEADS * (SEQ / 64), 128, 0, stream>>>(Qb, Kb, Vtb, Ob);
  k_oproj<<<dim3(MROWS / 128, EMB / 64), 128, 0, stream>>>(Ob, wto, bo, out);
}
